// GraphAttentionLayer_50921132261796
// MI455X (gfx1250) — compile-verified, matrix operand order checked by reading, not run on hardware
//
#include <hip/hip_runtime.h>

typedef __attribute__((ext_vector_type(16))) _Float16 v16h;
typedef __attribute__((ext_vector_type(8)))  _Float16 v8h;
typedef __attribute__((ext_vector_type(4)))  _Float16 v4h;
typedef __attribute__((ext_vector_type(8)))  float    v8f;

union H16 { v16h v; v8h h[2]; };

#define NROWS 8192
#define FIN   512
#define FOUT  256
#define LOG2E 1.44269504088896f

#if __has_builtin(__builtin_amdgcn_exp2f)
#define EXP2F(x) __builtin_amdgcn_exp2f(x)
#else
#define EXP2F(x) exp2f(x)
#endif

// ---------------------------------------------------------------------------
// K0: WT[n][k] = (f16) W[k][n]   (256 x 512 f16, B-operand friendly layout)
// ---------------------------------------------------------------------------
__global__ __launch_bounds__(256) void prep_wt_k(const float* __restrict__ W,
                                                 _Float16* __restrict__ WT) {
  int idx = blockIdx.x * 256 + threadIdx.x;   // 0 .. 131071
  int n = idx >> 9;
  int k = idx & 511;
  WT[idx] = (_Float16)W[k * FOUT + n];
}

// ---------------------------------------------------------------------------
// K1: Wh = h @ W + bW  via v_wmma_f32_16x16x32_f16.
// 4 waves per 16-row block; each wave owns 4 n-tiles (64 cols) -> 32 acc
// VGPRs, all 4 B tiles live at once (batched loads).
// Output written TRANSPOSED as f16: WhT[n][i].
// ---------------------------------------------------------------------------
__global__ __launch_bounds__(128) void gemm_wh_k(const float* __restrict__ h,
                                                 const _Float16* __restrict__ WT,
                                                 const float* __restrict__ bW,
                                                 _Float16* __restrict__ WhT) {
  const int  lane = threadIdx.x & 31;
  const int  wsub = threadIdx.x >> 5;        // 0..3 : n-panel
  const int  r0   = blockIdx.x * 16;
  const bool lo   = lane < 16;
  const int  m    = lane & 15;
  const int  sel  = lo ? 0 : 8;

  const float*    hrow  = h + (size_t)(r0 + m) * FIN;
  const _Float16* wbase = WT + (size_t)(wsub * 64 + m) * FIN + sel;

  v8f acc[4];
#pragma unroll
  for (int tt = 0; tt < 4; ++tt) {
    float bwv = bW[wsub * 64 + tt * 16 + m];
#pragma unroll
    for (int r = 0; r < 8; ++r) acc[tt][r] = bwv;
  }

#pragma unroll 2
  for (int ks = 0; ks < 16; ++ks) {          // K = 512 in steps of 32
    const int k0 = ks * 32;
    const float* hp = hrow + k0 + sel;
    float4 q0 = *(const float4*)(hp);
    float4 q1 = *(const float4*)(hp + 4);
    float4 q2 = *(const float4*)(hp + 16);
    float4 q3 = *(const float4*)(hp + 20);
    v16h a;
    a[0]=(_Float16)q0.x;  a[1]=(_Float16)q0.y;  a[2]=(_Float16)q0.z;  a[3]=(_Float16)q0.w;
    a[4]=(_Float16)q1.x;  a[5]=(_Float16)q1.y;  a[6]=(_Float16)q1.z;  a[7]=(_Float16)q1.w;
    a[8]=(_Float16)q2.x;  a[9]=(_Float16)q2.y;  a[10]=(_Float16)q2.z; a[11]=(_Float16)q2.w;
    a[12]=(_Float16)q3.x; a[13]=(_Float16)q3.y; a[14]=(_Float16)q3.z; a[15]=(_Float16)q3.w;
#pragma unroll
    for (int tt = 0; tt < 4; ++tt) {
      const _Float16* bp = wbase + (size_t)(tt * 16) * FIN + k0;
      H16 b;
      b.h[0] = *(const v8h*)bp;
      b.h[1] = *(const v8h*)(bp + 16);
      acc[tt] = __builtin_amdgcn_wmma_f32_16x16x32_f16(
          false, a, false, b.v, (short)0, acc[tt], false, false);
    }
  }

  const int i0 = r0 + sel;
#pragma unroll
  for (int tt = 0; tt < 4; ++tt) {
    const int n = wsub * 64 + tt * 16 + m;
    v8h ph;
#pragma unroll
    for (int r = 0; r < 8; ++r) ph[r] = (_Float16)acc[tt][r];
    *(v8h*)(WhT + ((size_t)n << 13) + i0) = ph;
  }
}

// ---------------------------------------------------------------------------
// K2: f1[i] = Wh[i]·a1, f2b[i] = Wh[i]·a2 + ba   (coalesced reads of WhT)
// ---------------------------------------------------------------------------
__global__ __launch_bounds__(256) void f1f2_k(const _Float16* __restrict__ WhT,
                                              const float* __restrict__ a1,
                                              const float* __restrict__ a2,
                                              const float* __restrict__ bap,
                                              float* __restrict__ f1,
                                              float* __restrict__ f2b) {
  int i = blockIdx.x * 256 + threadIdx.x;
  float s1 = 0.f, s2 = 0.f;
  for (int n = 0; n < FOUT; ++n) {
    float w = (float)WhT[((size_t)n << 13) + i];
    s1 = fmaf(w, a1[n], s1);
    s2 = fmaf(w, a2[n], s2);
  }
  f1[i]  = s1;
  f2b[i] = s2 + *bap;
}

// ---------------------------------------------------------------------------
// K3: F2MAX = max_j f2b[j]   (single-block LDS reduction)
// ---------------------------------------------------------------------------
__global__ __launch_bounds__(1024) void f2max_k(const float* __restrict__ f2b,
                                                float* __restrict__ outmax) {
  __shared__ float red[1024];
  int tid = threadIdx.x;
  float v = -3.0e38f;
  for (int i = tid; i < NROWS; i += 1024) v = fmaxf(v, f2b[i]);
  red[tid] = v;
  __syncthreads();
  for (int s = 512; s > 0; s >>= 1) {
    if (tid < s) red[tid] = fmaxf(red[tid], red[tid + s]);
    __syncthreads();
  }
  if (tid == 0) *outmax = red[0];
}

// ---------------------------------------------------------------------------
// K4: out = softmax(mask(lr(f1_i + f2_j + ba))) @ Wh   (single pass,
// fixed per-row shift m_i = lr(f1_i + F2MAX) >= all row logits).
//
// Block = 256 threads (8 waves), 16 rows. j-split x2 (wave groups), n-split
// x4 inside a group (4 n-tiles -> 32 acc VGPRs per wave).
// P tile produced cooperatively (coalesced adj/f2), double-buffered in LDS.
// B tiles double-buffered in REGISTERS and loaded a full chunk ahead, so no
// L2 latency sits between the barrier and the WMMAs. One barrier per chunk.
// ---------------------------------------------------------------------------
#define PPAD 40   // padded pbuf row (halves) to spread LDS banks

__global__ __launch_bounds__(256) void attn_k(const int* __restrict__ adj,
                                              const _Float16* __restrict__ WhT,
                                              const float* __restrict__ f1,
                                              const float* __restrict__ f2b,
                                              const float* __restrict__ f2maxp,
                                              float* __restrict__ out) {
  const int  tid  = threadIdx.x;        // 0..255
  const int  lane = tid & 31;
  const int  wave = tid >> 5;           // 0..7
  const int  wgrp = wave >> 2;          // j-half this wave consumes
  const int  wsub = wave & 3;           // n-panel (64 cols)
  const int  r0   = blockIdx.x * 16;
  const bool lo   = lane < 16;
  const int  m    = lane & 15;
  const int  sel  = lo ? 0 : 8;

  // ---- producer role: this thread computes 4 P values of row pm, half g ----
  const int g    = tid >> 7;            // j-half this thread produces for
  const int t7   = tid & 127;
  const int pm   = t7 >> 3;             // row 0..15
  const int pj   = (t7 & 7) * 4;        // j offset inside 32-chunk
  const int prow = r0 + pm;

  const float F2MAX = *f2maxp;
  const float f1p   = f1[prow];
  const float tmx   = f1p + F2MAX;
  const float mexp  = fmaxf(tmx, 0.2f * tmx) * LOG2E;   // shift in log2 units

  const int*   adjP = adj + (size_t)prow * NROWS + (size_t)g * 4096 + pj;
  const float* f2P  = f2b + (size_t)g * 4096 + pj;

  // ---- consumer role pointers ----
  const _Float16* bbase =
      WhT + (size_t)(wsub * 64 + m) * NROWS + (size_t)wgrp * 4096 + sel;

  __shared__ _Float16 pbuf[2][2][16][PPAD];   // [j-half][parity][row][k]
  __shared__ float    accS[4][16][64];        // group-1 partials
  __shared__ float    lS[256];
  __shared__ float    ltotS[16];

  v8f acc[4];
#pragma unroll
  for (int tt = 0; tt < 4; ++tt)
#pragma unroll
    for (int r = 0; r < 8; ++r) acc[tt][r] = 0.f;
  float lsum = 0.f;

  auto pfun = [&](int av, float fv) -> float {
    float t  = f1p + fv;
    float lr = fmaxf(t, 0.2f * t);             // leaky_relu slope 0.2
    float p  = EXP2F(lr * LOG2E - mexp);       // exp(e - m) <= 1
    return (av > 0) ? p : 0.0f;                // adjacency mask
  };

  auto produceStore = [&](int4 av, float4 fv, int par) {
    float p0 = pfun(av.x, fv.x);
    float p1 = pfun(av.y, fv.y);
    float p2 = pfun(av.z, fv.z);
    float p3 = pfun(av.w, fv.w);
    lsum += (p0 + p1) + (p2 + p3);
    v4h pp;
    pp[0] = (_Float16)p0; pp[1] = (_Float16)p1;
    pp[2] = (_Float16)p2; pp[3] = (_Float16)p3;
    *(v4h*)&pbuf[g][par][pm][pj] = pp;
  };

  auto loadB = [&](H16* b, int c) {
    const _Float16* bp = bbase + (size_t)c * 32;
    b[0].h[0] = *(const v8h*)(bp);
    b[0].h[1] = *(const v8h*)(bp + 16);
    b[1].h[0] = *(const v8h*)(bp + 16 * NROWS);
    b[1].h[1] = *(const v8h*)(bp + 16 * NROWS + 16);
    b[2].h[0] = *(const v8h*)(bp + 32 * NROWS);
    b[2].h[1] = *(const v8h*)(bp + 32 * NROWS + 16);
    b[3].h[0] = *(const v8h*)(bp + 48 * NROWS);
    b[3].h[1] = *(const v8h*)(bp + 48 * NROWS + 16);
  };

  auto consume = [&](int par, H16* b) {
    const _Float16* pA = &pbuf[wgrp][par][m][sel];
    H16 a;
    a.h[0] = *(const v8h*)pA;
    a.h[1] = *(const v8h*)(pA + 16);
    acc[0] = __builtin_amdgcn_wmma_f32_16x16x32_f16(false, a.v, false, b[0].v,
                                                    (short)0, acc[0], false, false);
    acc[1] = __builtin_amdgcn_wmma_f32_16x16x32_f16(false, a.v, false, b[1].v,
                                                    (short)0, acc[1], false, false);
    acc[2] = __builtin_amdgcn_wmma_f32_16x16x32_f16(false, a.v, false, b[2].v,
                                                    (short)0, acc[2], false, false);
    acc[3] = __builtin_amdgcn_wmma_f32_16x16x32_f16(false, a.v, false, b[3].v,
                                                    (short)0, acc[3], false, false);
  };

  // ---- prologue: chunk 0 operands ----
  int4   adjv0 = *(const int4*)(adjP);
  float4 f2v0  = *(const float4*)(f2P);
  int4   adjv1;
  float4 f2v1;
  H16 bA[4], bB[4];
  loadB(bA, 0);

  for (int c = 0; c < 128; c += 2) {
    // ================= chunk c (parity 0) =================
    produceStore(adjv0, f2v0, 0);
    adjv1 = *(const int4*)(adjP + (size_t)(c + 1) * 32);
    f2v1  = *(const float4*)(f2P + (size_t)(c + 1) * 32);
    if (c + 8 < 128)
      __builtin_prefetch(adjP + (size_t)(c + 8) * 32, 0, 1);
    loadB(bB, c + 1);          // next chunk's B, a full chunk ahead
    __syncthreads();           // P[c] visible
    consume(0, bA);

    // ================= chunk c+1 (parity 1) =================
    produceStore(adjv1, f2v1, 1);
    if (c + 2 < 128) {
      adjv0 = *(const int4*)(adjP + (size_t)(c + 2) * 32);
      f2v0  = *(const float4*)(f2P + (size_t)(c + 2) * 32);
      loadB(bA, c + 2);
    }
    if (c + 9 < 128)
      __builtin_prefetch(adjP + (size_t)(c + 9) * 32, 0, 1);
    __syncthreads();           // P[c+1] visible
    consume(1, bB);
  }

  // ---- combine the two j-halves, normalize by l, write output ----
  lS[tid] = lsum;
  if (wgrp == 1) {
#pragma unroll
    for (int tt = 0; tt < 4; ++tt)
#pragma unroll
      for (int r = 0; r < 8; ++r) accS[wsub][sel + r][tt * 16 + m] = acc[tt][r];
  }
  __syncthreads();
  if (tid < 16) {
    float s = 0.f;
#pragma unroll
    for (int k = 0; k < 8; ++k) s += lS[tid * 8 + k] + lS[128 + tid * 8 + k];
    ltotS[tid] = s;
  }
  __syncthreads();
  if (wgrp == 0) {
#pragma unroll
    for (int tt = 0; tt < 4; ++tt)
#pragma unroll
      for (int r = 0; r < 8; ++r) {
        int M = sel + r;
        float v = acc[tt][r] + accS[wsub][M][tt * 16 + m];
        out[(size_t)(r0 + M) * FOUT + (wsub * 64 + tt * 16 + m)] = v / ltotS[M];
      }
  }
}

// ---------------------------------------------------------------------------
// Launcher. Inputs: h, adj, W, bW, a1, a2, ba.  Output: f32 [8192][256].
// Workspace (~4.6 MB): WhT f16 4MiB | WT f16 256KiB | f1 | f2b | f2max
// ---------------------------------------------------------------------------
extern "C" void kernel_launch(void* const* d_in, const int* in_sizes, int n_in,
                              void* d_out, int out_size, void* d_ws, size_t ws_size,
                              hipStream_t stream) {
  const float* h   = (const float*)d_in[0];
  const int*   adj = (const int*)d_in[1];
  const float* W   = (const float*)d_in[2];
  const float* bW  = (const float*)d_in[3];
  const float* a1  = (const float*)d_in[4];
  const float* a2  = (const float*)d_in[5];
  const float* ba  = (const float*)d_in[6];
  float* out = (float*)d_out;

  char* ws = (char*)d_ws;
  _Float16* WhT   = (_Float16*)ws;                              // 4 MiB
  _Float16* WT    = (_Float16*)(ws + (size_t)4 * 1024 * 1024);  // 256 KiB
  float*    f1    = (float*)(ws + (size_t)4 * 1024 * 1024 + 256 * 1024);
  float*    f2b   = f1 + NROWS;
  float*    f2max = f2b + NROWS;

  prep_wt_k<<<dim3(512), dim3(256), 0, stream>>>(W, WT);
  gemm_wh_k<<<dim3(NROWS / 16), dim3(128), 0, stream>>>(h, WT, bW, WhT);
  f1f2_k<<<dim3(NROWS / 256), dim3(256), 0, stream>>>(WhT, a1, a2, ba, f1, f2b);
  f2max_k<<<dim3(1), dim3(1024), 0, stream>>>(f2b, f2max);
  attn_k<<<dim3(NROWS / 16), dim3(256), 0, stream>>>(adj, WhT, f1, f2b, f2max, out);
}
